// SparseEdgeUpdateLayer_21225728377480
// MI455X (gfx1250) — matrix operand order repaired and checked by reading, not compiled, not run on hardware
//
#include <hip/hip_runtime.h>
#include <hip/hip_bf16.h>

typedef _Float16 half_t;
typedef __attribute__((ext_vector_type(16))) _Float16 v16h;
typedef __attribute__((ext_vector_type(8)))  _Float16 v8h;
typedef __attribute__((ext_vector_type(4)))  _Float16 v4h;
typedef __attribute__((ext_vector_type(8)))  float    v8f;

#define IN_DIM   384
#define NODE_DIM 128
#define OUT_DIM  128
#define MTILE    64          // edges per block
#define LDA      392         // f16 row stride for x / hn tile (pad to break bank aliasing)
#define LDH      388         // f32 row stride for h tile
#define SMEM_BYTES (512 + MTILE*LDA*2 + MTILE*LDH*4)   // 512 + 50176 + 99328 = 150016

// ---------------- weight conversion: f32 -> f16, transposed -----------------
// w1t[o*384 + k] = W1[k*384 + o]   (o in [0,384))
// w2t[o*384 + k] = W2[k*128 + o]   (o in [0,128))
__global__ __launch_bounds__(256) void convert_weights_kernel(
    const float* __restrict__ W1, const float* __restrict__ W2,
    half_t* __restrict__ w1t, half_t* __restrict__ w2t)
{
    int i = blockIdx.x * 256 + threadIdx.x;
    const int n1 = IN_DIM * IN_DIM;          // 147456
    const int n2 = OUT_DIM * IN_DIM;         // 49152
    if (i < n1) {
        int o = i / IN_DIM, k = i - o * IN_DIM;
        w1t[i] = (half_t)W1[k * IN_DIM + o];
    } else if (i < n1 + n2) {
        int j = i - n1;
        int o = j / IN_DIM, k = j - o * IN_DIM;
        w2t[j] = (half_t)W2[k * OUT_DIM + o];
    }
}

// ---------------- fragment loaders (per CDNA5 ISA §7.12.2 layouts) ----------
// A (16x32 f16): lanes 0-15 row m, K{0..7,16..23}; lanes 16-31 row m, K{8..15,24..31}
__device__ __forceinline__ v16h load_a_frag(const half_t* base, int rowBase,
                                            int k, int m, int hi)
{
    const half_t* p = base + (rowBase + m) * LDA + k * 32 + 8 * hi;
    union { v16h v; v8h c[2]; } u;
    u.c[0] = *(const v8h*)p;
    u.c[1] = *(const v8h*)(p + 16);
    return u.v;
}
// B (32x16 f16): lane n = column; lanes 0-15 K{0..7,16..23}; lanes 16-31 K{8..15,24..31}
__device__ __forceinline__ v16h load_b_frag(const half_t* __restrict__ WT,
                                            int nt, int k, int m, int hi)
{
    const half_t* p = WT + (size_t)(16 * nt + m) * IN_DIM + k * 32 + 8 * hi;
    union { v16h v; v8h c[2]; } u;
    u.c[0] = *(const v8h*)p;
    u.c[1] = *(const v8h*)(p + 16);
    return u.v;
}

__device__ __forceinline__ v8f wmma_f16(v16h a, v16h b, v8f c)
{
    return __builtin_amdgcn_wmma_f32_16x16x32_f16(
        /*neg_a=*/false, a, /*neg_b=*/false, b,
        /*c_mod=*/(short)0, c, /*reuse_a=*/false, /*reuse_b=*/false);
}

// ---------------- fused edge-MLP kernel -------------------------------------
__global__ __launch_bounds__(256) void edge_mlp_kernel(
    const float* __restrict__ node_feats,   // [N,128]
    const float* __restrict__ edge_feats,   // [E,128]
    const int*   __restrict__ edge_index,   // [2,E]
    const half_t* __restrict__ w1t,         // [384,384] (out-major)
    const half_t* __restrict__ w2t,         // [128,384] (out-major)
    const float* __restrict__ b1,
    const float* __restrict__ gamma,
    const float* __restrict__ beta,
    const float* __restrict__ b2,
    float* __restrict__ out,                // [E,128]
    int E)
{
    extern __shared__ char smem[];
    int*    s_i = (int*)smem;                                   // 64
    int*    s_j = s_i + MTILE;                                  // 64
    half_t* s_x = (half_t*)(smem + 512);                        // [64][392] f16 (x, later hn)
    float*  s_h = (float*)(smem + 512 + MTILE * LDA * 2);       // [64][388] f32

    const int tid  = threadIdx.x;
    const int lane = tid & 31;
    const int wave = tid >> 5;
    const int m    = lane & 15;
    const int hi   = lane >> 4;
    const long long e0 = (long long)blockIdx.x * MTILE;

    // ---- stage 0: edge indices into LDS ----
    if (tid < MTILE) {
        long long e = e0 + tid;
        if (e >= E) e = E - 1;
        s_i[tid] = edge_index[e];
        s_j[tid] = edge_index[(long long)E + e];
    }
    __syncthreads();

    // ---- stage 1: gather + f32->f16 into LDS [64][384] ----
    #pragma unroll
    for (int it = 0; it < (MTILE * IN_DIM / 4) / 256; ++it) {   // 24 iters
        int q   = tid + 256 * it;          // [0,6144)
        int row = q / 96;
        int col = (q - row * 96) * 4;
        const float* src;
        if (col < 128) {
            src = node_feats + (size_t)s_i[row] * NODE_DIM + col;
        } else if (col < 256) {
            src = node_feats + (size_t)s_j[row] * NODE_DIM + (col - 128);
        } else {
            long long er = e0 + row; if (er >= E) er = E - 1;
            src = edge_feats + (size_t)er * NODE_DIM + (col - 256);
        }
        float4 v = *(const float4*)src;
        v4h hv = { (half_t)v.x, (half_t)v.y, (half_t)v.z, (half_t)v.w };
        *(v4h*)(s_x + row * LDA + col) = hv;
    }
    __syncthreads();

    // ---- stage 2: GEMM1  h[64,384] = x @ W1 + b1  (f16 WMMA, f32 acc) ----
    for (int mp = 0; mp < 2; ++mp) {                 // pairs of M-tiles
        v8f acc[2][3] = {};
        for (int k = 0; k < IN_DIM / 32; ++k) {      // 12 K-steps
            v16h a0 = load_a_frag(s_x, (2 * mp + 0) * 16, k, m, hi);
            v16h a1 = load_a_frag(s_x, (2 * mp + 1) * 16, k, m, hi);
            #pragma unroll
            for (int j = 0; j < 3; ++j) {
                v16h b = load_b_frag(w1t, wave + 8 * j, k, m, hi);
                acc[0][j] = wmma_f16(a0, b, acc[0][j]);
                acc[1][j] = wmma_f16(a1, b, acc[1][j]);
            }
        }
        #pragma unroll
        for (int j = 0; j < 3; ++j) {
            int nt = wave + 8 * j;
            float b1v = b1[16 * nt + m];
            #pragma unroll
            for (int mm = 0; mm < 2; ++mm) {
                int rowb = (2 * mp + mm) * 16 + 8 * hi;
                #pragma unroll
                for (int r = 0; r < 8; ++r)
                    s_h[(rowb + r) * LDH + 16 * nt + m] = acc[mm][j][r] + b1v;
            }
        }
    }
    __syncthreads();

    // ---- stage 3: LayerNorm + ReLU -> f16 hn (overwrites s_x) ----
    {
        float2 gv[6], bv[6];
        #pragma unroll
        for (int i = 0; i < 6; ++i) {
            gv[i] = *(const float2*)(gamma + 2 * lane + 64 * i);
            bv[i] = *(const float2*)(beta  + 2 * lane + 64 * i);
        }
        for (int rr = 0; rr < 8; ++rr) {
            int row = 8 * wave + rr;
            float2 hv[6];
            float s = 0.f, sq = 0.f;
            #pragma unroll
            for (int i = 0; i < 6; ++i) {
                hv[i] = *(const float2*)(s_h + row * LDH + 2 * lane + 64 * i);
                s  += hv[i].x + hv[i].y;
                sq += hv[i].x * hv[i].x + hv[i].y * hv[i].y;
            }
            #pragma unroll
            for (int off = 16; off >= 1; off >>= 1) {   // wave32 butterfly
                s  += __shfl_xor(s,  off, 32);
                sq += __shfl_xor(sq, off, 32);
            }
            float mu  = s  * (1.0f / IN_DIM);
            float var = sq * (1.0f / IN_DIM) - mu * mu;
            float rs  = rsqrtf(var + 1e-5f);
            #pragma unroll
            for (int i = 0; i < 6; ++i) {
                float y0 = (hv[i].x - mu) * rs * gv[i].x + bv[i].x;
                float y1 = (hv[i].y - mu) * rs * gv[i].y + bv[i].y;
                y0 = fmaxf(y0, 0.f); y1 = fmaxf(y1, 0.f);
                union { _Float16 h[2]; unsigned u; } pk;
                pk.h[0] = (_Float16)y0; pk.h[1] = (_Float16)y1;
                *(unsigned*)(s_x + row * LDA + 2 * lane + 64 * i) = pk.u;
            }
        }
    }
    __syncthreads();

    // ---- stage 4: GEMM2  out[64,128] = hn @ W2 + b2 ----
    {
        int mt  = wave >> 1;            // 0..3
        int ntb = (wave & 1) * 4;       // 0 or 4
        v8f acc[4] = {};
        for (int k = 0; k < IN_DIM / 32; ++k) {
            v16h a = load_a_frag(s_x, mt * 16, k, m, hi);
            #pragma unroll
            for (int j = 0; j < 4; ++j) {
                v16h b = load_b_frag(w2t, ntb + j, k, m, hi);
                acc[j] = wmma_f16(a, b, acc[j]);
            }
        }
        #pragma unroll
        for (int j = 0; j < 4; ++j) {
            int nt = ntb + j;
            float b2v = b2[16 * nt + m];
            long long rowg = e0 + mt * 16 + 8 * hi;
            #pragma unroll
            for (int r = 0; r < 8; ++r) {
                long long rr = rowg + r;
                if (rr < E)
                    out[rr * OUT_DIM + 16 * nt + m] = acc[j][r] + b2v;
            }
        }
    }
}

// ---------------- host launcher ---------------------------------------------
extern "C" void kernel_launch(void* const* d_in, const int* in_sizes, int n_in,
                              void* d_out, int out_size, void* d_ws, size_t ws_size,
                              hipStream_t stream)
{
    const float* node_feats = (const float*)d_in[0];
    const float* edge_feats = (const float*)d_in[1];
    const int*   edge_index = (const int*)d_in[2];   // jnp.int64 w/o x64 -> int32
    const float* W1    = (const float*)d_in[3];
    const float* b1    = (const float*)d_in[4];
    const float* gamma = (const float*)d_in[5];
    const float* beta  = (const float*)d_in[6];
    const float* W2    = (const float*)d_in[7];
    const float* b2    = (const float*)d_in[8];
    float* out = (float*)d_out;

    const int E = in_sizes[1] / NODE_DIM;            // edge_feats is [E,128]

    half_t* w1t = (half_t*)d_ws;                                    // 384*384 f16
    half_t* w2t = (half_t*)((char*)d_ws + (size_t)IN_DIM * IN_DIM * 2); // 128*384 f16

    const int nconv = IN_DIM * IN_DIM + OUT_DIM * IN_DIM;           // 196608
    convert_weights_kernel<<<(nconv + 255) / 256, 256, 0, stream>>>(W1, W2, w1t, w2t);

    const int nblocks = (E + MTILE - 1) / MTILE;                    // 12500
    edge_mlp_kernel<<<nblocks, 256, SMEM_BYTES, stream>>>(
        node_feats, edge_feats, edge_index, w1t, w2t,
        b1, gamma, beta, b2, out, E);
}
